// E3NN_50070728736816
// MI455X (gfx1250) — matrix operand order repaired and checked by reading, not compiled, not run on hardware
//
#include <hip/hip_runtime.h>
#include <math.h>

// ---------------------------------------------------------------------------
// CDNA5 (gfx1250) e3nn message-passing layer.
// GEMMs via v_wmma_f32_16x16x32_f16 (f16 in, f32 accumulate), wave32.
// Weights pre-packed into per-lane fragment blobs (32B vector loads).
// Edge kernel shares s[dst]/v[dst] gathers across message groups.
// ---------------------------------------------------------------------------

typedef __attribute__((ext_vector_type(16))) _Float16 v16h;
typedef __attribute__((ext_vector_type(8)))  _Float16 v8h;
typedef __attribute__((ext_vector_type(8)))  float    v8f;

#define WPB 8

static constexpr float K_INV_SQRT3   = 0.57735026918962576f;
static constexpr float K_INV_SQRT2   = 0.70710678118654752f;
static constexpr float K_INV_SQRTAVG = 0.25f;   // 1/sqrt(16) neighbor norm

__device__ __forceinline__ v8f wmma32(v16h a, v16h b, v8f c) {
  // 8 args: (neg_a, A, neg_b, B, c_mod, C, reuse_a, reuse_b)
  return __builtin_amdgcn_wmma_f32_16x16x32_f16(false, a, false, b, (short)0, c,
                                                false, false);
}

__device__ __forceinline__ float silu_f(float x) {
  return x / (1.0f + __expf(-x));
}

// ---------------------------------------------------------------------------
// Fragment-packed B weights: fragment f=(tn*tiles_k+tk) is 32 lanes x 16 halves,
// stored so lane l loads its whole v16h contiguously (32B, 2x b128, coalesced).
// Mapping matches ISA 7.12.2 B layout: n = lane&15, K = (e<8) ? (lane>>4)*8 + e : 16 + (lane>>4)*8 + (e-8).
__global__ void e3nn_pack_b(const float* __restrict__ W, v16h* __restrict__ P,
                            int Ktot, int nld, int tiles_k, int nfrags,
                            float scale) {
  int idx = blockIdx.x * blockDim.x + threadIdx.x;   // (frag, lane)
  if (idx >= nfrags * 32) return;
  int f    = idx >> 5;
  int lane = idx & 31;
  int tn   = f / tiles_k;
  int tk   = f - tn * tiles_k;
  int n    = lane & 15;
  int kh   = (lane >> 4) & 1;
  v16h b;
#pragma unroll
  for (int e = 0; e < 16; ++e) {
    int k = tk * 32 + ((e < 8) ? (kh * 8 + e) : (16 + kh * 8 + (e - 8)));
    float val = (k < Ktot) ? W[(size_t)k * nld + tn * 16 + n] * scale : 0.0f;
    b[e] = (_Float16)val;
  }
  P[idx] = b;
}

__device__ __forceinline__ v16h load_b_frag(const v16h* __restrict__ P,
                                            int tiles_k, int tn, int tk) {
  return P[((tn * tiles_k + tk) << 5) + (threadIdx.x & 31)];
}

// A-fragment (16x32 f16) from row-major f32, contiguous K. Row clamped (no
// zero-guard needed: D row m depends only on A row m; padded rows never stored).
__device__ __forceinline__ v16h load_a_rowmajor(const float* __restrict__ src,
                                                int row0, int ld, int col0,
                                                int nrows) {
  int lane = threadIdx.x & 31;
  int m    = lane & 15;
  int kh   = (lane >> 4) & 1;
  int r    = row0 + m;
  r        = (r < nrows) ? r : (nrows - 1);
  const float* p = src + (size_t)r * ld + col0 + kh * 8;
  const float4* q0 = (const float4*)p;         // K = kh*8 + 0..7
  const float4* q1 = (const float4*)(p + 16);  // K = 16 + kh*8 + 0..7
  float4 x0 = q0[0], x1 = q0[1], x2 = q1[0], x3 = q1[1];
  v16h a;
  a[0]  = (_Float16)x0.x; a[1]  = (_Float16)x0.y;
  a[2]  = (_Float16)x0.z; a[3]  = (_Float16)x0.w;
  a[4]  = (_Float16)x1.x; a[5]  = (_Float16)x1.y;
  a[6]  = (_Float16)x1.z; a[7]  = (_Float16)x1.w;
  a[8]  = (_Float16)x2.x; a[9]  = (_Float16)x2.y;
  a[10] = (_Float16)x2.z; a[11] = (_Float16)x2.w;
  a[12] = (_Float16)x3.x; a[13] = (_Float16)x3.y;
  a[14] = (_Float16)x3.z; a[15] = (_Float16)x3.w;
  return a;
}

// A-fragment with strided K (xyz-interleaved sources), row clamped.
__device__ __forceinline__ v16h load_a_strided(const float* __restrict__ src,
                                               int row0, int ld, int col0,
                                               int cstride, int nrows) {
  int lane = threadIdx.x & 31;
  int m    = lane & 15;
  int kh   = (lane >> 4) & 1;
  int r    = row0 + m;
  r        = (r < nrows) ? r : (nrows - 1);
  const float* p = src + (size_t)r * ld + col0;
  v16h a;
#pragma unroll
  for (int e = 0; e < 16; ++e) {
    int k = (e < 8) ? (kh * 8 + e) : (16 + kh * 8 + (e - 8));
    a[e] = (_Float16)p[k * cstride];
  }
  return a;
}

// A-fragment from per-wave LDS f16 tile (16 rows x 64 halves): 2x ds_load_b128.
__device__ __forceinline__ v16h load_a_lds(const _Float16* __restrict__ t16,
                                           int k0) {
  int lane = threadIdx.x & 31;
  int m    = lane & 15;
  int kh   = (lane >> 4) & 1;
  const _Float16* p = t16 + m * 64 + k0 + kh * 8;
  v8h lo = *(const v8h*)(p);
  v8h hi = *(const v8h*)(p + 16);
  return __builtin_shufflevector(lo, hi, 0, 1, 2, 3, 4, 5, 6, 7,
                                 8, 9, 10, 11, 12, 13, 14, 15);
}

// ---------------------------------------------------------------------------
// s = s_in @ W1_0/8 ; v[:,:,c] = v_in[:,:,c] @ W1_1/sqrt(32)
__global__ void __launch_bounds__(32 * WPB)
e3nn_node_pre(const float* __restrict__ nh, const v16h* __restrict__ W1_0p,
              const v16h* __restrict__ W1_1p, float* __restrict__ s_out,
              float* __restrict__ v_out, int N) {
  int wave = blockIdx.x * (blockDim.x >> 5) + (threadIdx.x >> 5);
  int row0 = wave * 16;
  if (row0 >= N) return;
  int lane = threadIdx.x & 31;
  int n    = lane & 15;
  int half = (lane >> 4) & 1;

  v16h a0 = load_a_rowmajor(nh, row0, 160, 0, N);
  v16h a1 = load_a_rowmajor(nh, row0, 160, 32, N);
#pragma unroll
  for (int t = 0; t < 4; ++t) {
    v8f c = {};
    c = wmma32(a0, load_b_frag(W1_0p, 2, t, 0), c);
    c = wmma32(a1, load_b_frag(W1_0p, 2, t, 1), c);
#pragma unroll
    for (int r = 0; r < 8; ++r)
      s_out[(size_t)(row0 + r + 8 * half) * 64 + t * 16 + n] = c[r];
  }

  // Buffer all 6 v C-fragments, then store x/y/z contiguously (b96-mergeable).
  v8f cvv[3][2];
#pragma unroll
  for (int comp = 0; comp < 3; ++comp) {
    v16h av = load_a_strided(nh, row0, 160, 64 + comp, 3, N);
#pragma unroll
    for (int t = 0; t < 2; ++t) {
      v8f c = {};
      c = wmma32(av, load_b_frag(W1_1p, 1, t, 0), c);
      cvv[comp][t] = c;
    }
  }
#pragma unroll
  for (int t = 0; t < 2; ++t)
#pragma unroll
    for (int r = 0; r < 8; ++r) {
      size_t base = (size_t)(row0 + r + 8 * half) * 96 + (t * 16 + n) * 3;
      v_out[base + 0] = cvv[0][t][r];
      v_out[base + 1] = cvv[1][t][r];
      v_out[base + 2] = cvv[2][t][r];
    }
}

// ---------------------------------------------------------------------------
// Fused per-edge MLP (WMMA) + equivariant messages + atomic scatter.
// All LDS tiles are wave-private; same-wave DS ops are in-order -> no barriers.
__global__ void __launch_bounds__(32 * WPB)
e3nn_edge(const float* __restrict__ edge_attr, const float* __restrict__ edge_sh,
          const int* __restrict__ edge_index, const float* __restrict__ s,
          const float* __restrict__ v, const v16h* __restrict__ Wfc1p,
          const v16h* __restrict__ Wfc2p, const v16h* __restrict__ Wfc3p,
          float* __restrict__ ns, float* __restrict__ nv, int E) {
  __shared__ __align__(16) _Float16 hbuf[WPB][16 * 64];
  __shared__ float shv[WPB][16][4];
  __shared__ int   esrc[WPB][16];
  __shared__ int   edst[WPB][16];

  int w    = threadIdx.x >> 5;
  int lane = threadIdx.x & 31;
  int n    = lane & 15;
  int half = (lane >> 4) & 1;
  int e0   = (blockIdx.x * WPB + w) * 16;
  if (e0 >= E) return;

  // Stage per-edge scalars; prefetch the gathered node rows.
  if (lane < 16) {
    int e = e0 + lane;
    int sidx = 0, didx = 0;
    float q0 = 0.f, q1 = 0.f, q2 = 0.f, q3 = 0.f;
    if (e < E) {
      sidx = edge_index[e];
      didx = edge_index[E + e];
      const float4 sh4 = *(const float4*)(edge_sh + (size_t)e * 4);
      q0 = sh4.x; q1 = sh4.y; q2 = sh4.z; q3 = sh4.w;
      __builtin_prefetch(s + (size_t)didx * 64, 0, 0);
      __builtin_prefetch(v + (size_t)didx * 96, 0, 0);
    }
    esrc[w][lane] = sidx;
    edst[w][lane] = didx;
    shv[w][lane][0] = q0; shv[w][lane][1] = q1;
    shv[w][lane][2] = q2; shv[w][lane][3] = q3;
  }

  // Layer-1 A fragment straight from edge_attr: K=8 valid, zero-padded to 32.
  v16h a_attr = {};
  if (half == 0) {
    int r = e0 + n;
    r = (r < E) ? r : (E - 1);
    const float4* q = (const float4*)(edge_attr + (size_t)r * 8);
    float4 x0 = q[0], x1 = q[1];
    a_attr[0] = (_Float16)x0.x; a_attr[1] = (_Float16)x0.y;
    a_attr[2] = (_Float16)x0.z; a_attr[3] = (_Float16)x0.w;
    a_attr[4] = (_Float16)x1.x; a_attr[5] = (_Float16)x1.y;
    a_attr[6] = (_Float16)x1.z; a_attr[7] = (_Float16)x1.w;
  }

  // Layer 1: h1 = silu(attr @ Wfc1/sqrt(8)) -> LDS tile (16x64 f16)
#pragma unroll
  for (int t = 0; t < 4; ++t) {
    v8f c = {};
    c = wmma32(a_attr, load_b_frag(Wfc1p, 1, t, 0), c);
#pragma unroll
    for (int r = 0; r < 8; ++r)
      hbuf[w][(r + 8 * half) * 64 + t * 16 + n] = (_Float16)silu_f(c[r]);
  }

  v16h a0 = load_a_lds(hbuf[w], 0);
  v16h a1 = load_a_lds(hbuf[w], 32);

  // Layer 2: h2 = silu(h1 @ Wfc2/8) -> LDS tile
  {
    v8f c2[4];
#pragma unroll
    for (int t = 0; t < 4; ++t) {
      v8f c = {};
      c = wmma32(a0, load_b_frag(Wfc2p, 2, t, 0), c);
      c = wmma32(a1, load_b_frag(Wfc2p, 2, t, 1), c);
      c2[t] = c;
    }
#pragma unroll
    for (int t = 0; t < 4; ++t)
#pragma unroll
      for (int r = 0; r < 8; ++r)
        hbuf[w][(r + 8 * half) * 64 + t * 16 + n] = (_Float16)silu_f(c2[t][r]);
  }

  a0 = load_a_lds(hbuf[w], 0);
  a1 = load_a_lds(hbuf[w], 32);

  // Hoist per-edge-slot scalars into registers (one LDS read each).
  int   srcs[8], dsts[8];
  float q0s[8], qxs[8], qys[8], qzs[8];
#pragma unroll
  for (int r = 0; r < 8; ++r) {
    int m = r + 8 * half;
    srcs[r] = esrc[w][m];
    dsts[r] = edst[w][m];
    q0s[r] = shv[w][m][0];
    qxs[r] = shv[w][m][1];
    qys[r] = shv[w][m][2];
    qzs[r] = shv[w][m][3];
  }

  // --- Scalar-channel groups: tile tp (w0 -> m0) and tp+4 (w1 -> mv0) share
  //     the s[dst] gather: one load feeds both messages.
#pragma unroll
  for (int tp = 0; tp < 4; ++tp) {
    v8f c0 = {}, c1 = {};
    c0 = wmma32(a0, load_b_frag(Wfc3p, 2, tp, 0), c0);
    c0 = wmma32(a1, load_b_frag(Wfc3p, 2, tp, 1), c0);
    c1 = wmma32(a0, load_b_frag(Wfc3p, 2, tp + 4, 0), c1);
    c1 = wmma32(a1, load_b_frag(Wfc3p, 2, tp + 4, 1), c1);
    int ch = tp * 16 + n;
#pragma unroll
    for (int r = 0; r < 8; ++r) {
      if (e0 + r + 8 * half >= E) continue;
      float sd = s[(size_t)dsts[r] * 64 + ch] * K_INV_SQRTAVG;
      float* nsp = ns + (size_t)srcs[r] * 96;
      float* nvp = nv + (size_t)srcs[r] * 384;
      atomicAdd(nsp + ch, c0[r] * sd * q0s[r]);        // m0
      float f = c1[r] * sd;                            // mv0
      atomicAdd(nvp + ch * 3 + 0, f * qxs[r]);
      atomicAdd(nvp + ch * 3 + 1, f * qys[r]);
      atomicAdd(nvp + ch * 3 + 2, f * qzs[r]);
    }
  }

  // --- Vector-channel groups: tiles 8+tp (w2 -> mv1), 10+tp (w3 -> m1),
  //     12+tp (w4 -> mv2) share the v[dst] gather: one 12B load, 7 messages.
#pragma unroll
  for (int tp = 0; tp < 2; ++tp) {
    v8f c2 = {}, c3 = {}, c4 = {};
    c2 = wmma32(a0, load_b_frag(Wfc3p, 2, 8 + tp, 0), c2);
    c2 = wmma32(a1, load_b_frag(Wfc3p, 2, 8 + tp, 1), c2);
    c3 = wmma32(a0, load_b_frag(Wfc3p, 2, 10 + tp, 0), c3);
    c3 = wmma32(a1, load_b_frag(Wfc3p, 2, 10 + tp, 1), c3);
    c4 = wmma32(a0, load_b_frag(Wfc3p, 2, 12 + tp, 0), c4);
    c4 = wmma32(a1, load_b_frag(Wfc3p, 2, 12 + tp, 1), c4);
    int ch = tp * 16 + n;
#pragma unroll
    for (int r = 0; r < 8; ++r) {
      if (e0 + r + 8 * half >= E) continue;
      const float* vp = v + (size_t)dsts[r] * 96 + ch * 3;
      float vx = vp[0], vy = vp[1], vz = vp[2];
      float qx = qxs[r], qy = qys[r], qz = qzs[r];
      float* nsp = ns + (size_t)srcs[r] * 96;
      float* nvp = nv + (size_t)srcs[r] * 384;
      // mv1 = w2 * vd * sh0
      float f2 = c2[r] * q0s[r] * K_INV_SQRTAVG;
      atomicAdd(nvp + (64 + ch) * 3 + 0, f2 * vx);
      atomicAdd(nvp + (64 + ch) * 3 + 1, f2 * vy);
      atomicAdd(nvp + (64 + ch) * 3 + 2, f2 * vz);
      // m1 = w3 * dot(vd, sh1) / sqrt3
      atomicAdd(nsp + 64 + ch, c3[r] * (vx * qx + vy * qy + vz * qz) *
                                   (K_INV_SQRT3 * K_INV_SQRTAVG));
      // mv2 = w4 * cross(vd, sh1) / sqrt2
      float f4 = c4[r] * (K_INV_SQRT2 * K_INV_SQRTAVG);
      atomicAdd(nvp + (96 + ch) * 3 + 0, f4 * (vy * qz - vz * qy));
      atomicAdd(nvp + (96 + ch) * 3 + 1, f4 * (vz * qx - vx * qz));
      atomicAdd(nvp + (96 + ch) * 3 + 2, f4 * (vx * qy - vy * qx));
    }
  }
}

// ---------------------------------------------------------------------------
// out_s = rmsnorm(ns@W2_0/sqrt(96) + s_in@Wsc0/8) * g0
// out_v = rmsnorm(nv.W2_1/sqrt(128) + v_in.Wsc1/sqrt(32)) * g1
__global__ void __launch_bounds__(32 * WPB)
e3nn_node_post(const float* __restrict__ nh, const float* __restrict__ ns,
               const float* __restrict__ nv, const v16h* __restrict__ W2_0p,
               const v16h* __restrict__ W2_1p, const v16h* __restrict__ Wsc0p,
               const v16h* __restrict__ Wsc1p, const float* __restrict__ g0,
               const float* __restrict__ g1, float* __restrict__ out, int N) {
  int wave = blockIdx.x * (blockDim.x >> 5) + (threadIdx.x >> 5);
  int row0 = wave * 16;
  if (row0 >= N) return;
  int lane = threadIdx.x & 31;
  int n    = lane & 15;
  int half = (lane >> 4) & 1;
  bool full = (row0 + 16) <= N;

  v16h as0 = load_a_rowmajor(ns, row0, 96, 0, N);
  v16h as1 = load_a_rowmajor(ns, row0, 96, 32, N);
  v16h as2 = load_a_rowmajor(ns, row0, 96, 64, N);
  v16h ak0 = load_a_rowmajor(nh, row0, 160, 0, N);
  v16h ak1 = load_a_rowmajor(nh, row0, 160, 32, N);

  v8f cs[4];
#pragma unroll
  for (int t = 0; t < 4; ++t) {
    v8f c = {};
    c = wmma32(as0, load_b_frag(W2_0p, 3, t, 0), c);
    c = wmma32(as1, load_b_frag(W2_0p, 3, t, 1), c);
    c = wmma32(as2, load_b_frag(W2_0p, 3, t, 2), c);
    c = wmma32(ak0, load_b_frag(Wsc0p, 2, t, 0), c);
    c = wmma32(ak1, load_b_frag(Wsc0p, 2, t, 1), c);
    cs[t] = c;
  }

  v8f cv[3][2];
#pragma unroll
  for (int comp = 0; comp < 3; ++comp) {
    v16h av0 = load_a_strided(nv, row0, 384, comp, 3, N);
    v16h av1 = load_a_strided(nv, row0, 384, comp + 96, 3, N);
    v16h av2 = load_a_strided(nv, row0, 384, comp + 192, 3, N);
    v16h av3 = load_a_strided(nv, row0, 384, comp + 288, 3, N);
    v16h avk = load_a_strided(nh, row0, 160, 64 + comp, 3, N);
#pragma unroll
    for (int t = 0; t < 2; ++t) {
      v8f c = {};
      c = wmma32(av0, load_b_frag(W2_1p, 4, t, 0), c);
      c = wmma32(av1, load_b_frag(W2_1p, 4, t, 1), c);
      c = wmma32(av2, load_b_frag(W2_1p, 4, t, 2), c);
      c = wmma32(av3, load_b_frag(W2_1p, 4, t, 3), c);
      c = wmma32(avk, load_b_frag(Wsc1p, 1, t, 0), c);
      cv[comp][t] = c;
    }
  }

  // RMS norms: row m = r + 8*half lives on the 16 lanes of this half.
#pragma unroll
  for (int r = 0; r < 8; ++r) {
    float ss = 0.0f;
#pragma unroll
    for (int t = 0; t < 4; ++t) ss += cs[t][r] * cs[t][r];
#pragma unroll
    for (int off = 1; off < 16; off <<= 1) ss += __shfl_xor(ss, off, 32);

    float sv = 0.0f;
#pragma unroll
    for (int comp = 0; comp < 3; ++comp)
#pragma unroll
      for (int t = 0; t < 2; ++t) sv += cv[comp][t][r] * cv[comp][t][r];
#pragma unroll
    for (int off = 1; off < 16; off <<= 1) sv += __shfl_xor(sv, off, 32);

    int node = row0 + r + 8 * half;
    if (full || node < N) {
      float irs = 1.0f / sqrtf(ss * (1.0f / 64.0f) + 1e-5f);
      float irv = 1.0f / sqrtf(sv * (1.0f / 32.0f) + 1e-5f);
#pragma unroll
      for (int t = 0; t < 4; ++t)
        out[(size_t)node * 160 + t * 16 + n] = cs[t][r] * irs * g0[t * 16 + n];
#pragma unroll
      for (int t = 0; t < 2; ++t) {
        float g = g1[t * 16 + n];
#pragma unroll
        for (int comp = 0; comp < 3; ++comp)
          out[(size_t)node * 160 + 64 + (t * 16 + n) * 3 + comp] =
              cv[comp][t][r] * irv * g;
      }
    }
  }
}

// ---------------------------------------------------------------------------
extern "C" void kernel_launch(void* const* d_in, const int* in_sizes, int n_in,
                              void* d_out, int out_size, void* d_ws, size_t ws_size,
                              hipStream_t stream) {
  const float* nh   = (const float*)d_in[0];
  const int*   eidx = (const int*)d_in[1];
  const float* esh  = (const float*)d_in[2];
  const float* eat  = (const float*)d_in[3];
  const float* W1_0 = (const float*)d_in[4];
  const float* W1_1 = (const float*)d_in[5];
  const float* Wfc1 = (const float*)d_in[6];
  const float* Wfc2 = (const float*)d_in[7];
  const float* Wfc3 = (const float*)d_in[8];
  const float* W2_0 = (const float*)d_in[9];
  const float* W2_1 = (const float*)d_in[10];
  const float* Wsc0 = (const float*)d_in[11];
  const float* Wsc1 = (const float*)d_in[12];
  const float* g0   = (const float*)d_in[13];
  const float* g1   = (const float*)d_in[14];

  const int N  = in_sizes[0] / 160;
  const int E  = in_sizes[1] / 2;
  const int Np = ((N + 15) / 16) * 16;   // row-padded workspace

  // Workspace: padded f32 activations/accumulators, then packed f16 fragments.
  float* ws  = (float*)d_ws;
  float* s_  = ws;
  float* v_  = s_ + (size_t)Np * 64;
  float* ns_ = v_ + (size_t)Np * 96;
  float* nv_ = ns_ + (size_t)Np * 96;
  v16h* pk    = (v16h*)(nv_ + (size_t)Np * 384);
  v16h* W1_0p = pk; pk += 8 * 32;    // 4 n-tiles x 2 k-steps
  v16h* W1_1p = pk; pk += 2 * 32;    // 2 x 1
  v16h* Wfc1p = pk; pk += 4 * 32;    // 4 x 1 (K=8 zero-padded)
  v16h* Wfc2p = pk; pk += 8 * 32;    // 4 x 2
  v16h* Wfc3p = pk; pk += 28 * 32;   // 14 x 2
  v16h* W2_0p = pk; pk += 12 * 32;   // 4 x 3
  v16h* W2_1p = pk; pk += 8 * 32;    // 2 x 4
  v16h* Wsc0p = pk; pk += 8 * 32;    // 4 x 2
  v16h* Wsc1p = pk; pk += 2 * 32;    // 2 x 1

  auto pack = [&](const float* src, v16h* dst, int Ktot, int nld, int tiles_k,
                  int nfrags, float sc) {
    e3nn_pack_b<<<(nfrags * 32 + 127) / 128, 128, 0, stream>>>(
        src, dst, Ktot, nld, tiles_k, nfrags, sc);
  };
  pack(W1_0, W1_0p, 64, 64, 2, 8, 0.125f);                  // 1/sqrt(64)
  pack(W1_1, W1_1p, 32, 32, 1, 2, 0.17677669529663689f);    // 1/sqrt(32)
  pack(Wfc1, Wfc1p, 8, 64, 1, 4, 0.35355339059327373f);     // 1/sqrt(8)
  pack(Wfc2, Wfc2p, 64, 64, 2, 8, 0.125f);
  pack(Wfc3, Wfc3p, 64, 224, 2, 28, 0.125f);
  pack(W2_0, W2_0p, 96, 64, 3, 12, 0.10206207261596575f);   // 1/sqrt(96)
  pack(W2_1, W2_1p, 128, 32, 4, 8, 0.088388347648318447f);  // 1/sqrt(128)
  pack(Wsc0, Wsc0p, 64, 64, 2, 8, 0.125f);
  pack(Wsc1, Wsc1p, 32, 32, 1, 2, 0.17677669529663689f);

  // Zero scatter accumulators (ns | nv contiguous) each call.
  hipMemsetAsync(ns_, 0, (size_t)Np * (96 + 384) * sizeof(float), stream);

  const int ntiles  = (N + 15) / 16;
  const int nblocks = (ntiles + WPB - 1) / WPB;
  e3nn_node_pre<<<nblocks, 32 * WPB, 0, stream>>>(nh, W1_0p, W1_1p, s_, v_, N);

  const int etiles  = (E + 15) / 16;
  const int eblocks = (etiles + WPB - 1) / WPB;
  e3nn_edge<<<eblocks, 32 * WPB, 0, stream>>>(eat, esh, eidx, s_, v_, Wfc1p,
                                              Wfc2p, Wfc3p, ns_, nv_, E);

  e3nn_node_post<<<nblocks, 32 * WPB, 0, stream>>>(nh, ns_, nv_, W2_0p, W2_1p,
                                                   Wsc0p, Wsc1p, g0, g1,
                                                   (float*)d_out, N);
}
